// TRM_19198503813910
// MI455X (gfx1250) — matrix operand order repaired and checked by reading, not compiled, not run on hardware
//
#include <hip/hip_runtime.h>
#include <hip/hip_bf16.h>
#include <stdint.h>

// Problem constants (from reference setup_inputs)
#define B_   32
#define NS_  16
#define NP_  8
#define C_   256
#define T_   128
#define M_   (T_ * T_)      // 16384
#define AR_  144            // 128 phrase rows (s*8+p) + 16 sentence rows
#define NTILE 64            // columns of the video matrix per workgroup
#define SIGK 10.0f
#define WRES 0.5f

typedef __attribute__((ext_vector_type(16))) __bf16 v16bf;
typedef __attribute__((ext_vector_type(8)))  float  v8f;

union BF16x16 {
    v16bf    v;
    uint32_t u[8];
};

#if __has_builtin(__builtin_amdgcn_cvt_pk_bf16_f32)
__device__ __forceinline__ uint32_t pack2_bf16(float a, float b) {
    typedef __attribute__((ext_vector_type(2))) __bf16 v2bf;
    union { v2bf v; uint32_t u; } cv;
    cv.v = __builtin_amdgcn_cvt_pk_bf16_f32(a, b);
    return cv.u;
}
#else
__device__ __forceinline__ uint32_t pack2_bf16(float a, float b) {
    uint32_t ua = __float_as_uint(a);
    uint32_t ub = __float_as_uint(b);
    ua = (ua + 0x7FFFu + ((ua >> 16) & 1u)) >> 16;   // RNE
    ub = (ub + 0x7FFFu + ((ub >> 16) & 1u)) >> 16;
    return (ub << 16) | (ua & 0xFFFFu);
}
#endif

__device__ __forceinline__ float fast_sigmoid10(float x) {
    return 1.0f / (1.0f + __expf(-SIGK * x));
}

// ---------------------------------------------------------------------------
// Kernel 1: build packed bf16 A matrix: per batch 144 rows x 256 cols.
// Rows 0..127  : phrase_feat rows, index r = s*NP + p, L2-normalized over C
// Rows 128..143: sent_feat rows, L2-normalized over C
// ---------------------------------------------------------------------------
__global__ __launch_bounds__(256) void trm_prep(
    const float* __restrict__ sent,      // (B, NS, C)
    const float* __restrict__ phrase,    // (B, NS, NP, C)
    uint16_t* __restrict__ Abf)          // (B, AR_, C) bf16
{
    __shared__ float red[256];
    const int idx = blockIdx.x;
    const int b = idx / AR_;
    const int r = idx % AR_;
    const int c = threadIdx.x;

    const float* src = (r < NS_ * NP_)
        ? (phrase + ((size_t)b * NS_ * NP_ + r) * C_)
        : (sent   + ((size_t)b * NS_ + (r - NS_ * NP_)) * C_);

    float x = src[c];
    red[c] = x * x;
    __syncthreads();
    #pragma unroll
    for (int off = 128; off > 0; off >>= 1) {
        if (c < off) red[c] += red[c + off];
        __syncthreads();
    }
    const float scale = 1.0f / fmaxf(sqrtf(red[0]), 1e-12f);

    uint32_t u = __float_as_uint(x * scale);
    u = (u + 0x7FFFu + ((u >> 16) & 1u)) >> 16;
    Abf[((size_t)b * AR_ + r) * C_ + c] = (uint16_t)u;
}

// ---------------------------------------------------------------------------
// Kernel 2: fused scores. One workgroup (4 waves) per (batch, 64-col tile).
// Each wave owns 16 columns. Video data is fed to the WMMAs *unnormalized*;
// per-column sum-of-squares is accumulated during the fragment loads and the
// L2 scale (constant per lane in the D layout) is applied to the accumulators
// afterwards. No LDS staging, no hot-path barriers.
// ---------------------------------------------------------------------------
__global__ __launch_bounds__(128) void trm_main(
    const float* __restrict__ map2d,     // (B, C, T, T)
    const float* __restrict__ pweight,   // (B, NS, NP)
    const float* __restrict__ pmask,     // (B, NS, NP)
    const uint16_t* __restrict__ Abf,    // (B, AR_, C) bf16
    float* __restrict__ out_iou,         // (B, NS, T, T)
    float* __restrict__ out_pmap)        // (B, NS, NP, T, T)
{
    __shared__ float phatt[NS_][NTILE];

    const int b    = blockIdx.y;
    const int m0   = blockIdx.x * NTILE;
    const int tid  = threadIdx.x;
    const int lane = tid & 31;
    const int wave = tid >> 5;
    const int half = lane >> 4;          // 0: lanes 0-15, 1: lanes 16-31
    const int l16  = lane & 15;

    const int colLocal = wave * 16 + l16;        // this lane's column 0..63
    const int m = m0 + colLocal;

    const float*    vb = map2d + (size_t)b * C_ * M_ + m;   // column base
    const uint16_t* Ab = Abf + (size_t)b * AR_ * C_;

    v8f   acc[9] = {};       // 9 row tiles: 8 phrase sentence-pairs + 1 sent
    float ss = 0.0f;         // partial sum of squares for this column

    #pragma unroll 1
    for (int k = 0; k < 8; ++k) {                 // K = 256 in steps of 32
        // B fragment: 16 consecutive K (= C) values per lane.
        // lanes 0-15: K = 32k..32k+15 | lanes 16-31: K = 32k+16..32k+31
        const int c0 = 32 * k + half * 16;
        float x[16];
        #pragma unroll
        for (int i = 0; i < 16; ++i)
            x[i] = vb[(size_t)(c0 + i) * M_];
        BF16x16 bf;
        #pragma unroll
        for (int i = 0; i < 8; ++i) {
            ss += x[2 * i] * x[2 * i] + x[2 * i + 1] * x[2 * i + 1];
            bf.u[i] = pack2_bf16(x[2 * i], x[2 * i + 1]);
        }
        #pragma unroll
        for (int t = 0; t < 9; ++t) {
            // A fragment: row = 16t + l16; two contiguous 16B chunks at
            // K-offsets 32k + half*16 and 32k + half*16 + 8.
            const uint16_t* arow = Ab + (size_t)(16 * t + l16) * C_ + 32 * k + half * 16;
            BF16x16 af;
            const uint4 lo = *(const uint4*)(arow);
            const uint4 hi = *(const uint4*)(arow + 8);
            af.u[0] = lo.x; af.u[1] = lo.y; af.u[2] = lo.z; af.u[3] = lo.w;
            af.u[4] = hi.x; af.u[5] = hi.y; af.u[6] = hi.z; af.u[7] = hi.w;
            acc[t] = __builtin_amdgcn_wmma_f32_16x16x32_bf16(
                false, af.v, false, bf.v, (short)0, acc[t], false, false);
        }
    }

    // Combine the two half-column partials (lanes l and l^16 share a column),
    // then the per-column normalization scale for every element in this lane.
    ss += __shfl_xor(ss, 16, 32);
    const float cs = 1.0f / fmaxf(sqrtf(ss), 1e-12f);

    // ---- Epilogues ----
    const int mi = m >> 7;        // proposal start index
    const int mj = m & (T_ - 1);  // proposal end index
    const float mask2d = (mj >= mi) ? 1.0f : 0.0f;

    float*       pmapB = out_pmap + (size_t)b * NS_ * NP_ * M_;
    const float* wB    = pweight + b * NS_ * NP_;
    const float* mkB   = pmask   + b * NS_ * NP_;

    // Phrase tiles t=0..7: D VGPR g holds row (M=g | g+8) -> (s = 2t+half, p = g).
    #pragma unroll
    for (int t = 0; t < 8; ++t) {
        const int s = 2 * t + half;
        float att = 0.0f;
        #pragma unroll
        for (int g = 0; g < 8; ++g) {
            const float sc = acc[t][g] * cs;
            att += wB[s * NP_ + g] * sc;
            const float v = mask2d * mkB[s * NP_ + g] * fast_sigmoid10(sc);
            pmapB[(size_t)(s * NP_ + g) * M_ + m] = v;
        }
        phatt[s][colLocal] = att;   // produced & consumed by this wave only
    }
    __syncthreads();

    // Sentence tile t=8: VGPR g holds s = g + 8*half.
    float* iouB = out_iou + (size_t)b * NS_ * M_;
    #pragma unroll
    for (int g = 0; g < 8; ++g) {
        const int s = g + 8 * half;
        const float v = acc[8][g] * cs + WRES * phatt[s][colLocal];
        iouB[(size_t)s * M_ + m] = mask2d * fast_sigmoid10(v);
    }
}

// ---------------------------------------------------------------------------
extern "C" void kernel_launch(void* const* d_in, const int* in_sizes, int n_in,
                              void* d_out, int out_size, void* d_ws, size_t ws_size,
                              hipStream_t stream) {
    (void)in_sizes; (void)n_in; (void)out_size; (void)ws_size;
    const float* map2d  = (const float*)d_in[0];  // (B,C,T,T)
    const float* sent   = (const float*)d_in[1];  // (B,NS,C)
    const float* phrase = (const float*)d_in[2];  // (B,NS,NP,C)
    const float* pw     = (const float*)d_in[3];  // (B,NS,NP)
    const float* pm     = (const float*)d_in[4];  // (B,NS,NP)

    uint16_t* Abf = (uint16_t*)d_ws;              // B*144*256 bf16 = 2.36 MB

    float* out_iou  = (float*)d_out;                          // (B,NS,T,T)
    float* out_pmap = out_iou + (size_t)B_ * NS_ * M_;        // (B,NS,NP,T,T)

    trm_prep<<<B_ * AR_, 256, 0, stream>>>(sent, phrase, Abf);

    dim3 grid(M_ / NTILE, B_);
    trm_main<<<grid, 128, 0, stream>>>(map2d, pw, pm, Abf, out_iou, out_pmap);
}
